// PPO_52390011076771
// MI455X (gfx1250) — matrix operand order repaired and checked by reading, not compiled, not run on hardware
//
#include <hip/hip_runtime.h>
#include <hip/hip_bf16.h>

typedef __attribute__((ext_vector_type(16))) __bf16 v16bf;
typedef __attribute__((ext_vector_type(8)))  __bf16 v8bf;
typedef __attribute__((ext_vector_type(8)))  float  v8f;

namespace {
constexpr int   kT      = 262144;
constexpr int   kS      = 128;
constexpr int   kA      = 8;
constexpr int   kH      = 64;
constexpr float kGamma  = 0.98f;
constexpr float kEps    = 0.1f;

constexpr int ROWS_PB = 128;      // rows per block (8 waves x 16 rows)
constexpr int NWAVE   = 8;
constexpr int XS      = kS + 8;   // lds_x row stride (bf16), 272B = mult of 16
constexpr int W1S     = kS + 8;   // transposed-W1 k-stride
constexpr int W2S     = kH + 8;   // 144B = mult of 16
constexpr int HS      = kH + 8;   // hidden bounce row stride
}

__device__ __forceinline__ v8f wmma_bf16(v16bf a, v16bf b, v8f c) {
  // D(16x16,f32) = A(16x32,bf16) x B(32x16,bf16) + C
  return __builtin_amdgcn_wmma_f32_16x16x32_bf16(
      /*neg_a=*/false, a, /*neg_b=*/false, b,
      /*c_mod=*/(short)0, c, /*reuse_a=*/false, /*reuse_b=*/false);
}

// Build a 16-element bf16 fragment from two 16B-aligned 8-element LDS runs.
__device__ __forceinline__ v16bf load_frag16(const __bf16* p0, const __bf16* p1) {
  v8bf lo = *(const v8bf*)p0;
  v8bf hi = *(const v8bf*)p1;
  return __builtin_shufflevector(lo, hi, 0,1,2,3,4,5,6,7,8,9,10,11,12,13,14,15);
}

// One wave: 16 rows through swish(swish(X@W1+b1)@W2+b2) @ [Wa|Wc|0] + [ba|bc|0]
// xt: 16 x kS bf16 (stride XS); h: 16 x kH bf16 bounce (stride HS);
// ow: 16x16 f32 output tile; w1/w2/w3 transposed ([col][k]) bf16 in LDS.
__device__ __forceinline__ void mlp_tile(
    const __bf16* xt, __bf16* h, float* ow,
    const __bf16* w1, const __bf16* w2, const __bf16* w3,
    const float* b1, const float* b2, const float* b3, int lane) {
  const int half = lane >> 4;   // 0: K runs [0..7],[16..23]; 1: [8..15],[24..31]
  const int l    = lane & 15;

  // ---- A fragments of X (4 chunks of K=32) ----
  v16bf ax[4];
#pragma unroll
  for (int c = 0; c < 4; ++c) {
    const __bf16* p = xt + l * XS + c * 32 + half * 8;
    ax[c] = load_frag16(p, p + 16);
  }

  // ---- layer 1: H1 = swish(X@W1 + b1), N=64 in 4 tiles ----
#pragma unroll
  for (int n = 0; n < 4; ++n) {
    v8f acc;
#pragma unroll
    for (int i = 0; i < 8; ++i) acc[i] = 0.f;
#pragma unroll
    for (int c = 0; c < 4; ++c) {
      const __bf16* p = w1 + (n * 16 + l) * W1S + c * 32 + half * 8;
      acc = wmma_bf16(ax[c], load_frag16(p, p + 16), acc);
    }
    const float bias = b1[n * 16 + l];
#pragma unroll
    for (int v = 0; v < 8; ++v) {
      float x  = acc[v] + bias;
      float sw = x / (1.f + __expf(-x));           // swish
      h[(v + half * 8) * HS + n * 16 + l] = (__bf16)sw;
    }
  }

  // ---- layer 2: H2 = swish(H1@W2 + b2) ----
  v16bf ah[2];
#pragma unroll
  for (int c = 0; c < 2; ++c) {                    // preload so h can be reused
    const __bf16* p = h + l * HS + c * 32 + half * 8;
    ah[c] = load_frag16(p, p + 16);
  }
#pragma unroll
  for (int n = 0; n < 4; ++n) {
    v8f acc;
#pragma unroll
    for (int i = 0; i < 8; ++i) acc[i] = 0.f;
#pragma unroll
    for (int c = 0; c < 2; ++c) {
      const __bf16* p = w2 + (n * 16 + l) * W2S + c * 32 + half * 8;
      acc = wmma_bf16(ah[c], load_frag16(p, p + 16), acc);
    }
    const float bias = b2[n * 16 + l];
#pragma unroll
    for (int v = 0; v < 8; ++v) {
      float x  = acc[v] + bias;
      float sw = x / (1.f + __expf(-x));
      h[(v + half * 8) * HS + n * 16 + l] = (__bf16)sw;
    }
  }

  // ---- layer 3: [logits | v | 0...] = H2 @ Wcomb + bcomb (single N tile) ----
  v16bf a3[2];
#pragma unroll
  for (int c = 0; c < 2; ++c) {
    const __bf16* p = h + l * HS + c * 32 + half * 8;
    a3[c] = load_frag16(p, p + 16);
  }
  v8f acc;
#pragma unroll
  for (int i = 0; i < 8; ++i) acc[i] = 0.f;
#pragma unroll
  for (int c = 0; c < 2; ++c) {
    const __bf16* p = w3 + l * W2S + c * 32 + half * 8;
    acc = wmma_bf16(a3[c], load_frag16(p, p + 16), acc);
  }
  const float bias = b3[l];
#pragma unroll
  for (int v = 0; v < 8; ++v)
    ow[(v + half * 8) * 16 + l] = acc[v] + bias;
}

extern "C" __global__ __launch_bounds__(256)
void ppo_main(const float* __restrict__ s, const float* __restrict__ sp,
              const float* __restrict__ r, const float* __restrict__ done,
              const float* __restrict__ prob_a, const int* __restrict__ a,
              const float* __restrict__ W1, const float* __restrict__ b1,
              const float* __restrict__ W2, const float* __restrict__ b2,
              const float* __restrict__ Wa, const float* __restrict__ ba,
              const float* __restrict__ Wc, const float* __restrict__ bc,
              float* __restrict__ out, float* __restrict__ partials) {
  __shared__ __align__(16) __bf16 lds_x[ROWS_PB * XS];       // 128x128 bf16 tile
  __shared__ __align__(16) __bf16 sW1t[kH * W1S];            // [col][k]
  __shared__ __align__(16) __bf16 sW2t[kH * W2S];
  __shared__ __align__(16) __bf16 sW3t[16 * W2S];            // [Wa|Wc|0] transposed
  __shared__ float sb1[kH], sb2[kH], sb3[16];
  __shared__ __align__(16) __bf16 lds_h[NWAVE * 16 * HS];    // per-wave bounce
  __shared__ float lds_out[NWAVE * 256];                     // per-wave 16x16
  __shared__ float hub_s[ROWS_PB];

  const int tid      = threadIdx.x;
  const int w        = tid >> 5;
  const int lane     = tid & 31;
  const int blockRow = blockIdx.x * ROWS_PB;

  // ---- stage weights (transposed, bf16) + biases into LDS ----
  for (int e = tid; e < kH * kS; e += 256) {                 // W1: 8192
    int col = e & (kH - 1), k = e >> 6;
    sW1t[col * W1S + k] = (__bf16)W1[k * kH + col];
  }
  for (int e = tid; e < kH * kH; e += 256) {                 // W2: 4096
    int col = e & (kH - 1), k = e >> 6;
    sW2t[col * W2S + k] = (__bf16)W2[k * kH + col];
  }
  for (int e = tid; e < 16 * kH; e += 256) {                 // Wcomb: 1024
    int col = e & 15, k = e >> 4;
    float v = 0.f;
    if (col < kA)       v = Wa[k * kA + col];
    else if (col == kA) v = Wc[k];
    sW3t[col * W2S + k] = (__bf16)v;
  }
  if (tid < kH) { sb1[tid] = b1[tid]; sb2[tid] = b2[tid]; }
  if (tid < 16) sb3[tid] = (tid < kA) ? ba[tid] : (tid == kA ? bc[0] : 0.f);

  float pi_a = 0.f, v_s = 0.f, v_sp = 0.f;

  for (int pass = 0; pass < 2; ++pass) {
    const float* X = pass ? sp : s;
    __syncthreads();                                         // weights ready / tile free
    // ---- coalesced fp32 tile load -> bf16 LDS ----
    const float4* src = (const float4*)(X + (size_t)blockRow * kS);
#pragma unroll
    for (int i = 0; i < 16; ++i) {
      int e4 = tid + i * 256;                                // 4096 float4 per tile
      float4 v4 = src[e4];
      int e = e4 * 4;
      __bf16* d = &lds_x[(e >> 7) * XS + (e & 127)];
      d[0] = (__bf16)v4.x; d[1] = (__bf16)v4.y;
      d[2] = (__bf16)v4.z; d[3] = (__bf16)v4.w;
    }
    __syncthreads();

    mlp_tile(lds_x + w * 16 * XS, lds_h + w * 16 * HS, lds_out + w * 256,
             sW1t, sW2t, sW3t, sb1, sb2, sb3, lane);

    if (lane < 16) {                                         // one row per lane
      const float* o = lds_out + w * 256 + lane * 16;
      if (pass == 0) {
        float m = o[0];
#pragma unroll
        for (int j = 1; j < kA; ++j) m = fmaxf(m, o[j]);
        float sum = 0.f;
#pragma unroll
        for (int j = 0; j < kA; ++j) sum += __expf(o[j] - m);
        int row = blockRow + w * 16 + lane;
        int ai  = a[row];
        pi_a = __expf(o[ai] - m) / sum;
        v_s  = o[kA];
      } else {
        v_sp = o[kA];
      }
    }
  }

  // ---- PPO arithmetic (advantage == delta; scan axis has size 1) ----
  if (lane < 16) {
    int row   = blockRow + w * 16 + lane;
    float rr  = r[row], dn = done[row], pa = prob_a[row];
    float td    = rr + kGamma * v_sp * dn;                   // td_target (masked)
    float delta = rr + kGamma * v_sp - v_s;                  // delta (unmasked)
    float adv   = delta;
    float ratio = pi_a / pa;                                 // exp(log pi_a - log pa)
    float s1    = ratio * adv;
    float cl    = fminf(fmaxf(ratio, 1.f - kEps), 1.f + kEps);
    float s2    = cl * adv;
    out[row]    = -fminf(s1, s2);                            // huber mean added later
    float err = td - v_s;
    float ae  = fabsf(err);
    hub_s[w * 16 + lane] = (ae <= 1.f) ? 0.5f * err * err : (ae - 0.5f);
  }
  __syncthreads();
  if (tid == 0) {                                            // deterministic partial
    float sum = 0.f;
    for (int i = 0; i < ROWS_PB; ++i) sum += hub_s[i];
    partials[blockIdx.x] = sum;
  }
}

extern "C" __global__ __launch_bounds__(256)
void ppo_reduce(const float* __restrict__ partials, float* __restrict__ hsum, int n) {
  __shared__ float sm[256];
  float s = 0.f;
  for (int i = threadIdx.x; i < n; i += 256) s += partials[i];
  sm[threadIdx.x] = s;
  __syncthreads();
  for (int k = 128; k > 0; k >>= 1) {
    if ((int)threadIdx.x < k) sm[threadIdx.x] += sm[threadIdx.x + k];
    __syncthreads();
  }
  if (threadIdx.x == 0) hsum[0] = sm[0] * (1.0f / (float)kT);
}

extern "C" __global__ __launch_bounds__(256)
void ppo_add(float* __restrict__ out, const float* __restrict__ hsum) {
  int i = blockIdx.x * blockDim.x + threadIdx.x;
  out[i] += hsum[0];
}

extern "C" void kernel_launch(void* const* d_in, const int* in_sizes, int n_in,
                              void* d_out, int out_size, void* d_ws, size_t ws_size,
                              hipStream_t stream) {
  const float* s      = (const float*)d_in[0];
  const float* sp     = (const float*)d_in[1];
  const float* r      = (const float*)d_in[2];
  const float* done   = (const float*)d_in[3];
  const float* prob_a = (const float*)d_in[4];
  const int*   a      = (const int*)d_in[5];
  const float* W1     = (const float*)d_in[6];
  const float* b1     = (const float*)d_in[7];
  const float* W2     = (const float*)d_in[8];
  const float* b2     = (const float*)d_in[9];
  const float* Wa     = (const float*)d_in[10];
  const float* ba     = (const float*)d_in[11];
  const float* Wc     = (const float*)d_in[12];
  const float* bc     = (const float*)d_in[13];

  float* out      = (float*)d_out;
  float* ws       = (float*)d_ws;
  float* hsum     = ws;        // ws[0]
  float* partials = ws + 1;    // ws[1 .. nblk]

  const int nblk = kT / ROWS_PB;   // 2048
  ppo_main<<<nblk, 256, 0, stream>>>(s, sp, r, done, prob_a, a,
                                     W1, b1, W2, b2, Wa, ba, Wc, bc,
                                     out, partials);
  ppo_reduce<<<1, 256, 0, stream>>>(partials, hsum, nblk);
  ppo_add<<<kT / 256, 256, 0, stream>>>(out, hsum);
}
